// ReducedV1NetCell_36369783062531
// MI455X (gfx1250) — matrix operand order repaired and checked by reading, not compiled, not run on hardware
//
#include <hip/hip_runtime.h>
#include <hip/hip_bf16.h>

typedef __bf16 bf16;
typedef __attribute__((ext_vector_type(16))) __bf16 v16bf;
typedef __attribute__((ext_vector_type(8)))  __bf16 bf16x8;
typedef __attribute__((ext_vector_type(4)))  __bf16 bf16x4;
typedef __attribute__((ext_vector_type(8)))  float  v8f;
typedef __attribute__((ext_vector_type(4)))  int    v4i;

#if defined(__has_builtin)
#if __has_builtin(__builtin_amdgcn_global_load_async_to_lds_b128) && \
    __has_builtin(__builtin_amdgcn_s_wait_asynccnt)
#define USE_ASYNC 1
#endif
#endif
#ifndef USE_ASYNC
#define USE_ASYNC 0
#endif

constexpr int   kB   = 32;
constexpr int   kCIN = 64;
constexpr int   kCH  = 64;
constexpr int   kH   = 128;
constexpr int   kW   = 128;
constexpr int   kP   = kH * kW;      // 16384
constexpr int   kG   = 3 * kCH;      // 192
constexpr float kEPS = 1e-5f;

constexpr size_t SZ_ACT = (size_t)kB * kCIN * kP;        // 33,554,432 elems
constexpr size_t OFF_XB    = 0;                          // bf16 x, NHWC
constexpr size_t OFF_HB    = OFF_XB + SZ_ACT * 2;        // bf16 h, NHWC
constexpr size_t OFF_WG    = OFF_HB + SZ_ACT * 2;        // packed gate weights
constexpr size_t OFF_WE    = OFF_WG + (size_t)12 * 4 * 512 * 2;
constexpr size_t OFF_WI    = OFF_WE + (size_t)4 * 25 * 2 * 512 * 2;
constexpr size_t OFF_GATES = (OFF_WI + (size_t)4 * 25 * 2 * 512 * 2 + 255) & ~(size_t)255;
constexpr size_t OFF_HOR   = OFF_GATES + (size_t)kB * kG * kP * 4;
constexpr size_t OFF_STATS = OFF_HOR + (size_t)kB * kCH * kP * 4;

// ---------------- NCHW fp32 -> NHWC bf16 (LDS-tiled transpose, coalesced) ----
__global__ void k_nchw2nhwc(const float* __restrict__ s, bf16* __restrict__ d) {
    __shared__ bf16 ls[64 * 68];                 // [c][px], pad 68 vs banks
    const int  tid = threadIdx.x;
    const long pixBase = (long)blockIdx.x * 64;
    const int  n   = (int)(pixBase / kP);
    const int  hw0 = (int)(pixBase % kP);
    for (int idx = tid; idx < 1024; idx += 256) {
        int c = idx >> 4, g = idx & 15;
        float4 v = *(const float4*)&s[((size_t)n * 64 + c) * kP + hw0 + g * 4];
        bf16x4 o;
        o.x = (bf16)v.x; o.y = (bf16)v.y; o.z = (bf16)v.z; o.w = (bf16)v.w;
        *(bf16x4*)&ls[c * 68 + g * 4] = o;
    }
    __syncthreads();
    for (int idx = tid; idx < 1024; idx += 256) {
        int px = idx >> 4, g = idx & 15;          // g: group of 4 channels
        bf16x4 o;
        o.x = ls[(g * 4 + 0) * 68 + px];
        o.y = ls[(g * 4 + 1) * 68 + px];
        o.z = ls[(g * 4 + 2) * 68 + px];
        o.w = ls[(g * 4 + 3) * 68 + px];
        *(bf16x4*)&d[((size_t)n * kP + hw0 + px) * 64 + g * 4] = o;
    }
}

// Pack [Wxg;Whg] (each 192x64) into WMMA A-fragment layout: tile = mt*4+kt
__global__ void k_pack_wg(const float* __restrict__ wxg, const float* __restrict__ whg,
                          bf16* __restrict__ out) {
    int t = blockIdx.x * blockDim.x + threadIdx.x;
    if (t >= 12 * 4 * 32) return;
    int lane = t & 31, tile = t >> 5;
    int kt = tile & 3, mt = tile >> 2;
    int m  = mt * 16 + (lane & 15);
    int kb = (lane >> 4) * 16;
    const float* w = (kt < 2) ? wxg : whg;
    int kt2 = kt & 1;
    for (int j = 0; j < 16; ++j) {
        int kl = kb + j;                          // A-frag K mapping
        out[(size_t)tile * 512 + lane * 16 + j] = (bf16)w[m * 64 + kt2 * 32 + kl];
    }
}

// Pack 5x5 OIHW (64,64,5,5) into A-fragments: tile = (mt*25+tap)*2+kt
__global__ void k_pack_wc(const float* __restrict__ w, bf16* __restrict__ out) {
    int t = blockIdx.x * blockDim.x + threadIdx.x;
    if (t >= 4 * 25 * 2 * 32) return;
    int lane = t & 31, tile = t >> 5;
    int kt  = tile & 1;
    int tap = (tile >> 1) % 25;
    int mt  = tile / 50;
    int m   = mt * 16 + (lane & 15);
    int kb  = (lane >> 4) * 16;
    for (int j = 0; j < 16; ++j) {
        int kl = kb + j;
        int k  = kt * 32 + kl;
        out[(size_t)tile * 512 + lane * 16 + j] = (bf16)w[(m * 64 + k) * 25 + tap];
    }
}

__global__ void k_zero(float* __restrict__ p) { p[threadIdx.x] = 0.0f; }

// ------------------------- gates (1x1 GEMM, K=128), no LDS: NHWC B-frags ----
__global__ __launch_bounds__(256, 2) void k_gates(
    const bf16* __restrict__ xN, const bf16* __restrict__ hN,
    const bf16* __restrict__ wg, const float* __restrict__ bxg,
    const float* __restrict__ bhg, float* __restrict__ gates) {
    const int  tid = threadIdx.x;
    const long pixBase = (long)blockIdx.x * 64;
    const int  n   = (int)(pixBase / kP);
    const int  hw0 = (int)(pixBase % kP);
    const int lane = tid & 31, wave = tid >> 5;
    const int nt = wave & 3, mh = wave >> 2;
    const int nl = lane & 15, kb2 = (lane >> 4) * 16;
    const size_t pix = (size_t)pixBase + nt * 16 + nl;   // n*P + hw
    v8f acc[6] = {};
    for (int kt = 0; kt < 4; ++kt) {
        const bf16* src = (kt < 2) ? xN : hN;
        v16bf b = *(const v16bf*)&src[pix * 64 + (kt & 1) * 32 + kb2];
#pragma unroll
        for (int mtl = 0; mtl < 6; ++mtl) {
            int mt = mh * 6 + mtl;
            v16bf a = ((const v16bf*)(wg + (size_t)(mt * 4 + kt) * 512))[lane];
            acc[mtl] = __builtin_amdgcn_wmma_f32_16x16x32_bf16(
                false, a, false, b, (short)0, acc[mtl], false, false);
        }
    }
    const int mrow = (lane >> 4) * 8;
#pragma unroll
    for (int mtl = 0; mtl < 6; ++mtl) {
        int mt = mh * 6 + mtl;
#pragma unroll
        for (int r = 0; r < 8; ++r) {
            int cout = mt * 16 + mrow + r;
            float v = acc[mtl][r] + bxg[cout] + bhg[cout];
            v = 1.0f / (1.0f + __expf(-v));               // sigmoid
            gates[((size_t)n * kG + cout) * kP + hw0 + nt * 16 + nl] = v;
        }
    }
}

// ----------------- 5x5 exc+inh conv pair, implicit GEMM, fused hor ----------
// block: (n, y, xseg) -> 64 px x 64 couts. 4 waves, wave = M-tile; pixel
// subtile loop inside so each A-fragment feeds 4 B-fragments (8 WMMAs).
constexpr int LWc = 68;
__global__ __launch_bounds__(128, 4) void k_hconv(
    const bf16* __restrict__ hN, const float* __restrict__ hf,
    const bf16* __restrict__ we, const bf16* __restrict__ wi,
    const float* __restrict__ bexc, const float* __restrict__ binh,
    float* __restrict__ hor) {
    __shared__ bf16 t[5 * LWc * 64];                 // [row][col][ch] 43,520 B
    const int tid  = threadIdx.x;
    const int bid  = blockIdx.x;
    const int xseg = bid & 1;
    const int y    = (bid >> 1) & (kH - 1);
    const int n    = bid >> 8;
    const int xb0  = xseg * 64;
    // haloed tile load: 5*68 pixel cells, 8 x 16B chunks of channels each
    for (int idx = tid; idx < 5 * LWc * 8; idx += 128) {
        int chunk = idx & 7;
        int cell  = idx >> 3;
        int r = cell / LWc, col = cell % LWc;
        int gy = y + r - 2, gx = xb0 + col - 2;
        bf16* dst = &t[cell * 64 + chunk * 8];
        if ((unsigned)gy < (unsigned)kH && (unsigned)gx < (unsigned)kW) {
            const bf16* src = hN + ((size_t)n * kP + gy * kW + gx) * 64 + chunk * 8;
#if USE_ASYNC
            __builtin_amdgcn_global_load_async_to_lds_b128(
                (__attribute__((address_space(1))) v4i*)src,
                (__attribute__((address_space(3))) v4i*)dst, 0, 0);
#else
            *(bf16x8*)dst = *(const bf16x8*)src;
#endif
        } else {
            bf16x8 z = {};
            *(bf16x8*)dst = z;
        }
    }
#if USE_ASYNC
    __builtin_amdgcn_s_wait_asynccnt(0);
#endif
    __syncthreads();
    const int lane = tid & 31;
    const int mt   = tid >> 5;                     // wave -> M tile 0..3
    const int nl   = lane & 15, kb2 = (lane >> 4) * 16;
    v8f ace[4] = {}, aci[4] = {};
    for (int tap = 0; tap < 25; ++tap) {
        const int ky = tap / 5, kx = tap % 5;
#pragma unroll
        for (int kt = 0; kt < 2; ++kt) {
            v16bf aE = ((const v16bf*)(we + (size_t)((mt * 25 + tap) * 2 + kt) * 512))[lane];
            v16bf aI = ((const v16bf*)(wi + (size_t)((mt * 25 + tap) * 2 + kt) * 512))[lane];
#pragma unroll
            for (int nt = 0; nt < 4; ++nt) {
                const int col = nt * 16 + nl + kx;
                v16bf b = *(const v16bf*)&t[(ky * LWc + col) * 64 + kt * 32 + kb2];
                ace[nt] = __builtin_amdgcn_wmma_f32_16x16x32_bf16(
                    false, aE, false, b, (short)0, ace[nt], false, false);
                aci[nt] = __builtin_amdgcn_wmma_f32_16x16x32_bf16(
                    false, aI, false, b, (short)0, aci[nt], false, false);
            }
        }
    }
    const int mrow  = (lane >> 4) * 8;
    const int coutB = mt * 16 + mrow;
#pragma unroll
    for (int nt = 0; nt < 4; ++nt) {
        int px = xb0 + nt * 16 + nl;
#pragma unroll
        for (int r = 0; r < 8; ++r) {
            int cout = coutB + r;
            size_t o = ((size_t)n * kCH + cout) * kP + y * kW + px;
            float exc = ace[nt][r] + bexc[cout];
            float inh = aci[nt][r] + binh[cout];
            hor[o] = (hf[o] + exc) / (1.0f + __expf(inh));  // sigmoid(-inh)*(h+exc)
        }
    }
}

// ------------------------------- c_next + per-batch sum / sumsq -------------
__global__ void k_cnext(const float* __restrict__ gates, const float* __restrict__ hor,
                        const float* __restrict__ cin, float* __restrict__ cnext,
                        float* __restrict__ stats) {
    __shared__ float s1[256], s2_[256];
    const long base  = (long)blockIdx.x * 1024;
    const int  n     = (int)(base / ((long)kCH * kP));
    const long nbase = (long)n * kCH * kP;
    const long goff  = (long)n * kG * kP;
    float s = 0.f, q = 0.f;
    for (int u = 0; u < 4; ++u) {
        long e = base + threadIdx.x + u * 256;
        long r = e - nbase;
        long c = r / kP, hw = r % kP;
        float iv = gates[goff + c * kP + hw];
        float fv = gates[goff + (64 + c) * kP + hw];
        float cn = fv * cin[e] + iv * hor[e];
        cnext[e] = cn;
        s += cn; q += cn * cn;
    }
    s1[threadIdx.x] = s; s2_[threadIdx.x] = q;
    __syncthreads();
    for (int off = 128; off > 0; off >>= 1) {
        if (threadIdx.x < off) {
            s1[threadIdx.x]  += s1[threadIdx.x + off];
            s2_[threadIdx.x] += s2_[threadIdx.x + off];
        }
        __syncthreads();
    }
    if (threadIdx.x == 0) {
        atomicAdd(&stats[2 * n],     s1[0]);
        atomicAdd(&stats[2 * n + 1], s2_[0]);
    }
}

// ------------------------------- h_next = o * tanh(LN(c')) ------------------
__global__ void k_hnext(const float* __restrict__ gates, const float* __restrict__ cnext,
                        const float* __restrict__ stats, float* __restrict__ hout) {
    const long base  = (long)blockIdx.x * 1024;
    const int  n     = (int)(base / ((long)kCH * kP));
    const float inv  = 1.0f / (float)((long)kCH * kP);
    const float mu   = stats[2 * n] * inv;
    const float var  = stats[2 * n + 1] * inv - mu * mu;
    const float rs   = rsqrtf(var + kEPS);
    const long goff  = (long)n * kG * kP;
    const long nbase = (long)n * kCH * kP;
    for (int u = 0; u < 4; ++u) {
        long e = base + threadIdx.x + u * 256;
        long r = e - nbase;
        long c = r / kP, hw = r % kP;
        float ov = gates[goff + (128 + c) * kP + hw];
        hout[e] = ov * tanhf((cnext[e] - mu) * rs);
    }
}

extern "C" void kernel_launch(void* const* d_in, const int* in_sizes, int n_in,
                              void* d_out, int out_size, void* d_ws, size_t ws_size,
                              hipStream_t stream) {
    (void)in_sizes; (void)n_in; (void)out_size; (void)ws_size;
    const float* x    = (const float*)d_in[0];
    const float* h    = (const float*)d_in[1];
    const float* c    = (const float*)d_in[2];
    const float* wxg  = (const float*)d_in[3];
    const float* bxg  = (const float*)d_in[4];
    const float* whg  = (const float*)d_in[5];
    const float* bhg  = (const float*)d_in[6];
    const float* wexc = (const float*)d_in[7];
    const float* bexc = (const float*)d_in[8];
    const float* winh = (const float*)d_in[9];
    const float* binh = (const float*)d_in[10];

    char*  ws    = (char*)d_ws;
    bf16*  xN    = (bf16*)(ws + OFF_XB);
    bf16*  hN    = (bf16*)(ws + OFF_HB);
    bf16*  wg    = (bf16*)(ws + OFF_WG);
    bf16*  we    = (bf16*)(ws + OFF_WE);
    bf16*  wi    = (bf16*)(ws + OFF_WI);
    float* gates = (float*)(ws + OFF_GATES);
    float* hor   = (float*)(ws + OFF_HOR);
    float* stats = (float*)(ws + OFF_STATS);

    float* hnext = (float*)d_out;
    float* cnext = hnext + (size_t)kB * kCH * kP;

    const int nPixBlocks = (int)((size_t)kB * kP / 64);   // 8192
    k_nchw2nhwc<<<nPixBlocks, 256, 0, stream>>>(x, xN);
    k_nchw2nhwc<<<nPixBlocks, 256, 0, stream>>>(h, hN);
    k_pack_wg<<<6, 256, 0, stream>>>(wxg, whg, wg);
    k_pack_wc<<<25, 256, 0, stream>>>(wexc, we);
    k_pack_wc<<<25, 256, 0, stream>>>(winh, wi);
    k_zero<<<1, 64, 0, stream>>>(stats);

    k_gates<<<nPixBlocks, 256, 0, stream>>>(xN, hN, wg, bxg, bhg, gates);
    k_hconv<<<kB * kH * 2, 128, 0, stream>>>(hN, h, we, wi, bexc, binh, hor);
    k_cnext<<<(int)((size_t)kB * kCH * kP / 1024), 256, 0, stream>>>(gates, hor, c, cnext, stats);
    k_hnext<<<(int)((size_t)kB * kCH * kP / 1024), 256, 0, stream>>>(gates, cnext, stats, hnext);
}
